// MultiHeadAttention_46102178955917
// MI455X (gfx1250) — matrix operand order repaired and checked by reading, not compiled, not run on hardware
//
#include <hip/hip_runtime.h>
#include <hip/hip_bf16.h>

// ---------------------------------------------------------------------------
// MHA block for MI455X (gfx1250): bf16 WMMA GEMMs + flash attention + LN
// B=2, S=2048, E=1024, H=16, D=64.  All GEMMs via v_wmma_f32_16x16x32_bf16.
// GEMM waves compute 32x64 register tiles (8 WMMA accs) -> ~21 FLOP/byte at
// the load interface; attention uses 64-key chunks to amortize softmax VALU.
// ---------------------------------------------------------------------------

typedef __bf16 bf16;
typedef __attribute__((ext_vector_type(16))) __bf16 v16bf;
typedef __attribute__((ext_vector_type(8)))  float  v8f;

#define BATCH 2
#define SEQ   2048
#define EMBED 1024
#define HEADS 16
#define HDIM  64
#define MTOT  (BATCH * SEQ)   // 4096 rows

union FragBF {
    v16bf v;
    uint4 q[2];
};

// A-fragment 16x32 (MxK), 16-bit, per ISA 7.12.2:
// lane l: m = l&15, half = l>>4
//   VGPR0-3 : K = 8*half + 0..7   (packed pairs)
//   VGPR4-7 : K = 16 + 8*half + 0..7
__device__ __forceinline__ v16bf load_a_frag(const bf16* A, int lda, int row0,
                                             int k0, int lane) {
    int m = row0 + (lane & 15);
    int half = lane >> 4;
    const bf16* p = A + (size_t)m * lda + k0 + 8 * half;
    FragBF f;
    f.q[0] = *(const uint4*)(p);
    f.q[1] = *(const uint4*)(p + 16);
    return f.v;
}

// B-fragment 32x16 (KxN), 16-bit; B stored as [N][K] row-major (K contiguous):
// lane l: n = l&15, half = l>>4
//   VGPR0-3 : K = 8*half + 0..7   (packed pairs)
//   VGPR4-7 : K = 16 + 8*half + 0..7
__device__ __forceinline__ v16bf load_b_frag(const bf16* B, int ldb, int col0,
                                             int k0, int lane) {
    int n = col0 + (lane & 15);
    int half = lane >> 4;
    const bf16* p = B + (size_t)n * ldb + k0 + 8 * half;
    FragBF f;
    f.q[0] = *(const uint4*)(p);
    f.q[1] = *(const uint4*)(p + 16);
    return f.v;
}

__device__ __forceinline__ v8f wmma_bf16(v16bf a, v16bf b, v8f c) {
    return __builtin_amdgcn_wmma_f32_16x16x32_bf16(
        /*neg_a=*/false, a, /*neg_b=*/false, b,
        /*c_mod=*/(short)0, c, /*reuse_a=*/false, /*reuse_b=*/false);
}

// ---------------------------------------------------------------------------
// Kernel 1: f32 -> bf16 conversion
// ---------------------------------------------------------------------------
__global__ void cvt_f32_to_bf16(const float* __restrict__ in,
                                bf16* __restrict__ out, int n) {
    int i = blockIdx.x * blockDim.x + threadIdx.x;
    if (i < n) out[i] = (bf16)in[i];
}

// ---------------------------------------------------------------------------
// Kernel 2: fused QKV projection.  C[m,e] = sum_k X[m,k] * W[e,k]
// blockIdx.z: 0 -> Q ([B,H,S,D]), 1 -> K ([B,H,S,D]), 2 -> V ([B,H,D,S])
// Block: 256 thr = 8 waves (2x4); block tile 64x256; wave tile 32x64.
// Grid: (MTOT/64, EMBED/256, 3)
// ---------------------------------------------------------------------------
__global__ __launch_bounds__(256) void gemm_qkv(const bf16* __restrict__ X,
                                                const bf16* __restrict__ Wq,
                                                const bf16* __restrict__ Wk,
                                                const bf16* __restrict__ Wv,
                                                bf16* __restrict__ Qout,
                                                bf16* __restrict__ Kout,
                                                bf16* __restrict__ Vtout) {
    const bf16* W;
    bf16* out;
    int mode;
    if (blockIdx.z == 0)      { W = Wq; out = Qout;  mode = 0; }
    else if (blockIdx.z == 1) { W = Wk; out = Kout;  mode = 0; }
    else                      { W = Wv; out = Vtout; mode = 1; }

    int lane = threadIdx.x & 31;
    int wave = threadIdx.x >> 5;
    int m0 = blockIdx.x * 64  + (wave >> 2) * 32;   // 2 wave-rows of 32
    int n0 = blockIdx.y * 256 + (wave & 3) * 64;    // 4 wave-cols of 64

    v8f c[2][4] = {};
    for (int k0 = 0; k0 < EMBED; k0 += 32) {
        v16bf a0 = load_a_frag(X, EMBED, m0,      k0, lane);
        v16bf a1 = load_a_frag(X, EMBED, m0 + 16, k0, lane);
        v16bf b0 = load_b_frag(W, EMBED, n0,      k0, lane);
        v16bf b1 = load_b_frag(W, EMBED, n0 + 16, k0, lane);
        v16bf b2 = load_b_frag(W, EMBED, n0 + 32, k0, lane);
        v16bf b3 = load_b_frag(W, EMBED, n0 + 48, k0, lane);
        c[0][0] = wmma_bf16(a0, b0, c[0][0]);
        c[0][1] = wmma_bf16(a0, b1, c[0][1]);
        c[0][2] = wmma_bf16(a0, b2, c[0][2]);
        c[0][3] = wmma_bf16(a0, b3, c[0][3]);
        c[1][0] = wmma_bf16(a1, b0, c[1][0]);
        c[1][1] = wmma_bf16(a1, b1, c[1][1]);
        c[1][2] = wmma_bf16(a1, b2, c[1][2]);
        c[1][3] = wmma_bf16(a1, b3, c[1][3]);
    }

    int half = lane >> 4, nn = lane & 15;
    for (int i = 0; i < 2; ++i) {
        for (int t = 0; t < 4; ++t) {
            int e = n0 + t * 16 + nn;
            int h = e >> 6, d = e & 63;
            for (int r = 0; r < 8; ++r) {
                int m = m0 + i * 16 + r + 8 * half;
                int b = m >> 11, s = m & (SEQ - 1);
                size_t addr;
                if (mode == 0)
                    addr = ((((size_t)b * HEADS + h) * SEQ) + s) * HDIM + d;
                else
                    addr = (((size_t)b * HEADS + h) * HDIM + d) * SEQ + s;
                out[addr] = (bf16)c[i][t][r];
            }
        }
    }
}

// ---------------------------------------------------------------------------
// Kernel 3: flash attention.  One wave per 16-row query tile, 64-key chunks.
// Q,K: [B,H,S,D] bf16;  Vt: [B,H,D,S] bf16;  out attn: [B,S,H*D] bf16.
// grid: x = SEQ/16/8 = 16 (8 q-tiles per block), y = B*H = 32.
// ---------------------------------------------------------------------------
__global__ __launch_bounds__(256) void attn_flash(const bf16* __restrict__ Q,
                                                  const bf16* __restrict__ K,
                                                  const bf16* __restrict__ Vt,
                                                  bf16* __restrict__ Aout) {
    __shared__ bf16 lds_p[8][16 * 64];  // per-wave P staging (C-layout -> A-layout)

    int lane = threadIdx.x & 31;
    int wave = threadIdx.x >> 5;
    int bh = blockIdx.y;                       // b*HEADS + h
    int q0 = (blockIdx.x * 8 + wave) * 16;     // query tile start within SEQ

    const bf16* Qp = Q  + (size_t)bh * SEQ * HDIM;
    const bf16* Kp = K  + (size_t)bh * SEQ * HDIM;
    const bf16* Vp = Vt + (size_t)bh * HDIM * SEQ;

    v8f o[4] = {};                 // output accum, 4 d-tiles of 16
    float mrow[8], lrow[8];
    for (int r = 0; r < 8; ++r) { mrow[r] = -1e30f; lrow[r] = 0.0f; }

    // Q tile fragments (K-dim = D = 64 -> two 32-wide A fragments), loop-invariant
    v16bf qa0 = load_a_frag(Qp, HDIM, q0, 0,  lane);
    v16bf qa1 = load_a_frag(Qp, HDIM, q0, 32, lane);

    const float scale = 0.125f;    // 1/sqrt(64)

    for (int s0 = 0; s0 < SEQ; s0 += 64) {
        // scores: 16 q x 64 keys, fp32 (4 tiles x 2 k-steps = 8 WMMA)
        v8f sc[4] = {};
        for (int t = 0; t < 4; ++t) {
            v16bf kb0 = load_b_frag(Kp, HDIM, s0 + t * 16, 0,  lane);
            v16bf kb1 = load_b_frag(Kp, HDIM, s0 + t * 16, 32, lane);
            sc[t] = wmma_bf16(qa0, kb0, sc[t]);
            sc[t] = wmma_bf16(qa1, kb1, sc[t]);
        }

        // online softmax; row m = r + 8*(lane>>4), cols across 16-lane half
        float alpha[8];
        for (int r = 0; r < 8; ++r) {
            float a0 = sc[0][r] * scale, a1 = sc[1][r] * scale;
            float a2 = sc[2][r] * scale, a3 = sc[3][r] * scale;
            float mx = fmaxf(fmaxf(a0, a1), fmaxf(a2, a3));
            for (int msk = 1; msk < 16; msk <<= 1)
                mx = fmaxf(mx, __shfl_xor(mx, msk, 32));
            float mnew = fmaxf(mrow[r], mx);
            alpha[r] = __expf(mrow[r] - mnew);
            mrow[r] = mnew;
            float e0 = __expf(a0 - mnew), e1 = __expf(a1 - mnew);
            float e2 = __expf(a2 - mnew), e3 = __expf(a3 - mnew);
            sc[0][r] = e0; sc[1][r] = e1; sc[2][r] = e2; sc[3][r] = e3;
            float rs = (e0 + e1) + (e2 + e3);
            for (int msk = 1; msk < 16; msk <<= 1)
                rs += __shfl_xor(rs, msk, 32);
            lrow[r] = lrow[r] * alpha[r] + rs;
        }
        for (int t = 0; t < 4; ++t)
            for (int r = 0; r < 8; ++r) o[t][r] *= alpha[r];

        // P (C-layout fp32) -> LDS bf16 [16 x 64], re-read as two A-fragments
        bf16* P = lds_p[wave];
        int half = lane >> 4, nn = lane & 15;
        for (int r = 0; r < 8; ++r) {
            int m = r + 8 * half;
            P[m * 64 + nn]      = (bf16)sc[0][r];
            P[m * 64 + 16 + nn] = (bf16)sc[1][r];
            P[m * 64 + 32 + nn] = (bf16)sc[2][r];
            P[m * 64 + 48 + nn] = (bf16)sc[3][r];
        }
        __syncthreads();
        v16bf pa0 = load_a_frag(P, 64, 0, 0,  lane);
        v16bf pa1 = load_a_frag(P, 64, 0, 32, lane);
        __syncthreads();

        // O += P @ V  (B-fragments from Vt: n = d, K-dim = s contiguous)
        for (int t = 0; t < 4; ++t) {
            v16bf vb0 = load_b_frag(Vp, SEQ, t * 16, s0,      lane);
            v16bf vb1 = load_b_frag(Vp, SEQ, t * 16, s0 + 32, lane);
            o[t] = wmma_bf16(pa0, vb0, o[t]);
            o[t] = wmma_bf16(pa1, vb1, o[t]);
        }
    }

    // epilogue: normalize and store as [B*S, E] bf16 (E index = h*64 + d)
    int half = lane >> 4, nn = lane & 15;
    int b = bh >> 4, h = bh & 15;
    for (int t = 0; t < 4; ++t)
        for (int r = 0; r < 8; ++r) {
            int s = q0 + r + 8 * half;
            float val = o[t][r] / lrow[r];
            size_t row = (size_t)b * SEQ + s;
            Aout[row * EMBED + h * HDIM + t * 16 + nn] = (bf16)val;
        }
}

// ---------------------------------------------------------------------------
// Kernel 4: out projection + residual.  Y[m,e] = sum_k A[m,k]*Wo[e,k] + X[m,e]
// Same 32x64 wave tiling as gemm_qkv.  Grid: (MTOT/64, EMBED/256)
// ---------------------------------------------------------------------------
__global__ __launch_bounds__(256) void gemm_out(const bf16* __restrict__ A,
                                                const bf16* __restrict__ W,
                                                const float* __restrict__ X,
                                                float* __restrict__ Y) {
    int lane = threadIdx.x & 31;
    int wave = threadIdx.x >> 5;
    int m0 = blockIdx.x * 64  + (wave >> 2) * 32;
    int n0 = blockIdx.y * 256 + (wave & 3) * 64;

    v8f c[2][4] = {};
    for (int k0 = 0; k0 < EMBED; k0 += 32) {
        v16bf a0 = load_a_frag(A, EMBED, m0,      k0, lane);
        v16bf a1 = load_a_frag(A, EMBED, m0 + 16, k0, lane);
        v16bf b0 = load_b_frag(W, EMBED, n0,      k0, lane);
        v16bf b1 = load_b_frag(W, EMBED, n0 + 16, k0, lane);
        v16bf b2 = load_b_frag(W, EMBED, n0 + 32, k0, lane);
        v16bf b3 = load_b_frag(W, EMBED, n0 + 48, k0, lane);
        c[0][0] = wmma_bf16(a0, b0, c[0][0]);
        c[0][1] = wmma_bf16(a0, b1, c[0][1]);
        c[0][2] = wmma_bf16(a0, b2, c[0][2]);
        c[0][3] = wmma_bf16(a0, b3, c[0][3]);
        c[1][0] = wmma_bf16(a1, b0, c[1][0]);
        c[1][1] = wmma_bf16(a1, b1, c[1][1]);
        c[1][2] = wmma_bf16(a1, b2, c[1][2]);
        c[1][3] = wmma_bf16(a1, b3, c[1][3]);
    }

    int half = lane >> 4, nn = lane & 15;
    for (int i = 0; i < 2; ++i)
        for (int t = 0; t < 4; ++t) {
            int e = n0 + t * 16 + nn;
            for (int r = 0; r < 8; ++r) {
                size_t m = m0 + i * 16 + r + 8 * half;
                Y[m * EMBED + e] = c[i][t][r] + X[m * EMBED + e];
            }
        }
}

// ---------------------------------------------------------------------------
// Kernel 5: in-place LayerNorm over rows of 1024
// ---------------------------------------------------------------------------
__global__ __launch_bounds__(256) void layernorm_rows(float* __restrict__ Y,
                                                      const float* __restrict__ gamma,
                                                      const float* __restrict__ beta) {
    __shared__ float rsum[8], rssq[8];
    int row = blockIdx.x;
    float* y = Y + (size_t)row * EMBED;
    int t = threadIdx.x;

    float v[4];
    float s = 0.0f, ss = 0.0f;
    for (int i = 0; i < 4; ++i) {
        float x = y[t + i * 256];
        v[i] = x;
        s += x; ss += x * x;
    }
    for (int msk = 1; msk < 32; msk <<= 1) {
        s  += __shfl_xor(s,  msk, 32);
        ss += __shfl_xor(ss, msk, 32);
    }
    int wave = t >> 5, lane = t & 31;
    if (lane == 0) { rsum[wave] = s; rssq[wave] = ss; }
    __syncthreads();
    if (t == 0) {
        float ts = 0.0f, tss = 0.0f;
        for (int i = 0; i < 8; ++i) { ts += rsum[i]; tss += rssq[i]; }
        rsum[0] = ts; rssq[0] = tss;
    }
    __syncthreads();
    float mean = rsum[0] * (1.0f / EMBED);
    float var  = rssq[0] * (1.0f / EMBED) - mean * mean;
    float inv  = rsqrtf(var + 1e-5f);
    for (int i = 0; i < 4; ++i) {
        int col = t + i * 256;
        y[col] = (v[i] - mean) * inv * gamma[col] + beta[col];
    }
}

// ---------------------------------------------------------------------------
// Host launcher
// ---------------------------------------------------------------------------
extern "C" void kernel_launch(void* const* d_in, const int* in_sizes, int n_in,
                              void* d_out, int out_size, void* d_ws, size_t ws_size,
                              hipStream_t stream) {
    const float* x     = (const float*)d_in[0];
    const float* Wq    = (const float*)d_in[1];
    const float* Wk    = (const float*)d_in[2];
    const float* Wv    = (const float*)d_in[3];
    const float* Wout  = (const float*)d_in[4];
    const float* gamma = (const float*)d_in[5];
    const float* beta  = (const float*)d_in[6];
    float* out = (float*)d_out;

    // workspace layout (bf16 buffers), total 48 MiB
    char* ws = (char*)d_ws;
    bf16* xh   = (bf16*)(ws);                                  // [4096,1024]  8 MiB
    bf16* wqh  = (bf16*)(ws + (8u  << 20));                    // [1024,1024]  2 MiB
    bf16* wkh  = (bf16*)(ws + (10u << 20));
    bf16* wvh  = (bf16*)(ws + (12u << 20));
    bf16* woh  = (bf16*)(ws + (14u << 20));
    bf16* Qh   = (bf16*)(ws + (16u << 20));                    // [B,H,S,D]    8 MiB
    bf16* Kh   = (bf16*)(ws + (24u << 20));                    // [B,H,S,D]    8 MiB
    bf16* Vt   = (bf16*)(ws + (32u << 20));                    // [B,H,D,S]    8 MiB
    bf16* attn = (bf16*)(ws + (40u << 20));                    // [4096,1024]  8 MiB

    // 1. convert inputs to bf16
    {
        int n = MTOT * EMBED;
        cvt_f32_to_bf16<<<(n + 255) / 256, 256, 0, stream>>>(x, xh, n);
        n = EMBED * EMBED;
        cvt_f32_to_bf16<<<(n + 255) / 256, 256, 0, stream>>>(Wq,   wqh, n);
        cvt_f32_to_bf16<<<(n + 255) / 256, 256, 0, stream>>>(Wk,   wkh, n);
        cvt_f32_to_bf16<<<(n + 255) / 256, 256, 0, stream>>>(Wv,   wvh, n);
        cvt_f32_to_bf16<<<(n + 255) / 256, 256, 0, stream>>>(Wout, woh, n);
    }

    // 2. fused Q/K/V projections (WMMA, 32x64 wave tiles)
    {
        dim3 grid(MTOT / 64, EMBED / 256, 3);
        gemm_qkv<<<grid, 256, 0, stream>>>(xh, wqh, wkh, wvh, Qh, Kh, Vt);
    }

    // 3. flash attention (WMMA scores + PV, 64-key chunks)
    {
        dim3 grid(SEQ / 16 / 8, BATCH * HEADS);
        attn_flash<<<grid, 256, 0, stream>>>(Qh, Kh, Vt, attn);
    }

    // 4. out projection + residual (WMMA) -> d_out (f32)
    {
        dim3 grid(MTOT / 64, EMBED / 256);
        gemm_out<<<grid, 256, 0, stream>>>(attn, woh, x, out);
    }

    // 5. LayerNorm in place on d_out
    layernorm_rows<<<MTOT, 256, 0, stream>>>(out, gamma, beta);
}
